// UltraEfficientRouter_35957466202216
// MI455X (gfx1250) — matrix operand order repaired and checked by reading, not compiled, not run on hardware
//
#include <hip/hip_runtime.h>
#include <hip/hip_bf16.h>

typedef __attribute__((ext_vector_type(16))) __bf16 v16bf;
typedef __attribute__((ext_vector_type(8)))  float  v8f;
typedef __attribute__((ext_vector_type(4)))  float  v4f;

#define C_IN   1024
#define BATCH  32
#define HOUT   16
#define HW_OUT 256          // 16*16
#define M_TOT  (BATCH*HW_OUT)   // 8192
#define RED    64
#define NE     8
#define BN_EPS 1e-5f

static __device__ __forceinline__ unsigned short f2bf_u16(float f) {
  unsigned u = __builtin_bit_cast(unsigned, f);
  unsigned r = (u + 0x7FFFu + ((u >> 16) & 1u)) >> 16;   // round-to-nearest-even
  return (unsigned short)r;
}
// fast SiLU: x * v_rcp_f32(1 + exp(-x)) -- avoids IEEE div fixup chains
static __device__ __forceinline__ float silu_f(float x) {
  return x * __builtin_amdgcn_rcpf(1.0f + __expf(-x));
}

// -------------------------------------------------------------------------
// Kernel 0: zero the small accumulator region of the workspace
// -------------------------------------------------------------------------
__global__ void init_zero_kernel(float* p, int n) {
  int i = blockIdx.x * 256 + threadIdx.x;
  if (i < n) p[i] = 0.0f;
}

// -------------------------------------------------------------------------
// Kernel 1: depthwise 3x3 stride-2 pad-1 conv.
//   Block = (batch b, group of 8 channels). Writes h transposed to
//   [m = b*256 + p, c] row-major so the GEMM reads unit-stride K.
//   Also accumulates per-channel sum / sum-of-squares for BatchNorm.
// -------------------------------------------------------------------------
__global__ void __launch_bounds__(256) dw_conv_kernel(
    const float* __restrict__ x, const float* __restrict__ dwk,
    float* __restrict__ h, float* __restrict__ ch_sum, float* __restrict__ ch_sq)
{
  __shared__ float xs[8][32][32];     // 32 KB
  __shared__ float wsh[8][9];
  __shared__ float s_sum[8], s_sq[8];

  const int blk = blockIdx.x;
  const int b   = blk >> 7;           // 128 channel groups per image
  const int c0  = (blk & 127) * 8;
  const int tid = threadIdx.x;

  if (tid < 8)  { s_sum[tid] = 0.0f; s_sq[tid] = 0.0f; }
  if (tid < 72) { wsh[tid / 9][tid % 9] = dwk[(size_t)(c0 + tid / 9) * 9 + (tid % 9)]; }

  // 8 contiguous channel planes: 8192 floats, fully coalesced
  const float* xb  = x + ((size_t)b * C_IN + c0) * 1024;
  float*       xsf = &xs[0][0][0];
  for (int i = tid; i < 8192 / 4; i += 256)
    ((v4f*)xsf)[i] = ((const v4f*)xb)[i];
  __syncthreads();

  const int py = tid >> 4, px = tid & 15;
  const int iy0 = 2 * py - 1, ix0 = 2 * px - 1;

  float hv[8];
  #pragma unroll
  for (int c = 0; c < 8; ++c) {
    float acc = 0.0f;
    #pragma unroll
    for (int ky = 0; ky < 3; ++ky) {
      int iy = iy0 + ky;
      if (iy < 0 || iy > 31) continue;
      #pragma unroll
      for (int kx = 0; kx < 3; ++kx) {
        int ix = ix0 + kx;
        if (ix < 0 || ix > 31) continue;
        acc += wsh[c][ky * 3 + kx] * xs[c][iy][ix];
      }
    }
    hv[c] = acc;
  }

  // transposed store: 32B contiguous per thread
  float* hp = h + (size_t)(b * HW_OUT + tid) * C_IN + c0;
  v4f lo = {hv[0], hv[1], hv[2], hv[3]};
  v4f hi = {hv[4], hv[5], hv[6], hv[7]};
  ((v4f*)hp)[0] = lo;
  ((v4f*)hp)[1] = hi;

  // BN stats: wave32 shuffle-reduce, then LDS, then 8 global atomics/block
  #pragma unroll
  for (int c = 0; c < 8; ++c) {
    float s = hv[c], q = hv[c] * hv[c];
    #pragma unroll
    for (int off = 16; off > 0; off >>= 1) {
      s += __shfl_down(s, off, 32);
      q += __shfl_down(q, off, 32);
    }
    if ((tid & 31) == 0) { atomicAdd(&s_sum[c], s); atomicAdd(&s_sq[c], q); }
  }
  __syncthreads();
  if (tid < 8) {
    atomicAdd(&ch_sum[c0 + tid], s_sum[tid]);
    atomicAdd(&ch_sq [c0 + tid], s_sq [tid]);
  }
}

// -------------------------------------------------------------------------
// Kernel 2: finalize BN -> per-channel scale/shift
// -------------------------------------------------------------------------
__global__ void bn_finalize_kernel(
    const float* __restrict__ ch_sum, const float* __restrict__ ch_sq,
    const float* __restrict__ gamma,  const float* __restrict__ beta,
    float* __restrict__ scale, float* __restrict__ shift)
{
  int c = blockIdx.x * 256 + threadIdx.x;
  if (c >= C_IN) return;
  const float inv_n = 1.0f / (float)M_TOT;
  float mu  = ch_sum[c] * inv_n;
  float var = ch_sq[c] * inv_n - mu * mu;
  float is  = rsqrtf(var + BN_EPS);
  float sc  = gamma[c] * is;
  scale[c] = sc;
  shift[c] = beta[c] - mu * sc;
}

// -------------------------------------------------------------------------
// Kernel 3: pack pw_w [RED, C] fp32 -> bf16 (RNE)
// -------------------------------------------------------------------------
__global__ void pack_w_kernel(const float* __restrict__ w, unsigned short* __restrict__ wbf) {
  int i = blockIdx.x * 256 + threadIdx.x;
  if (i < RED * C_IN) wbf[i] = f2bf_u16(w[i]);
}

// -------------------------------------------------------------------------
// Kernel 4: WMMA GEMM  [8192,1024]x[1024,64] with fused BN+SiLU on A-load,
//           fused SiLU + spatial-mean pooling on the epilogue.
//   8 waves/block; each wave: 16(M) x 64(N) strip, K loop step 32 (bf16 WMMA).
// -------------------------------------------------------------------------
__global__ void __launch_bounds__(256) gemm_pool_kernel(
    const float* __restrict__ h, const float* __restrict__ scale,
    const float* __restrict__ shift, const unsigned short* __restrict__ wbf,
    float* __restrict__ pooled)
{
  __shared__ float pool[RED];
  const int tid = threadIdx.x;
  if (tid < RED) pool[tid] = 0.0f;
  __syncthreads();

  const int wave = tid >> 5, lane = tid & 31;
  const int half = lane >> 4, l15 = lane & 15;
  const int m = blockIdx.x * 128 + wave * 16 + l15;
  const float* hrow = h + (size_t)m * C_IN;

  v8f acc0 = {0,0,0,0,0,0,0,0};
  v8f acc1 = {0,0,0,0,0,0,0,0};
  v8f acc2 = {0,0,0,0,0,0,0,0};
  v8f acc3 = {0,0,0,0,0,0,0,0};

  for (int k0 = 0; k0 < C_IN; k0 += 32) {
    // A fragment, contiguous K order matching B:
    //   lanes 0-15: K = k0..k0+15; lanes 16-31: K = k0+16..k0+31
    const int ka = k0 + (half ? 16 : 0);
    const int kb = k0 + (half ? 24 : 8);
    v8f xa = *(const v8f*)(hrow  + ka);
    v8f xb = *(const v8f*)(hrow  + kb);
    v8f sa = *(const v8f*)(scale + ka);
    v8f sb = *(const v8f*)(scale + kb);
    v8f ta = *(const v8f*)(shift + ka);
    v8f tb = *(const v8f*)(shift + kb);

    v16bf afrag;
    #pragma unroll
    for (int j = 0; j < 8; ++j) {
      float ya = silu_f(xa[j] * sa[j] + ta[j]);   // BN + SiLU fused
      float yb = silu_f(xb[j] * sb[j] + tb[j]);
      afrag[j]     = (__bf16)ya;                  // native cvt, RNE
      afrag[j + 8] = (__bf16)yb;
    }

    // B fragments: lane n=l15 holds column n; lanes 0-15 K=k0..k0+15,
    // lanes 16-31 K=k0+16..k0+31 -> one 32B contiguous load per tile
    const int kw = k0 + (half ? 16 : 0);
    v16bf b0 = *(const v16bf*)(wbf + (size_t)(0  + l15) * C_IN + kw);
    v16bf b1 = *(const v16bf*)(wbf + (size_t)(16 + l15) * C_IN + kw);
    v16bf b2 = *(const v16bf*)(wbf + (size_t)(32 + l15) * C_IN + kw);
    v16bf b3 = *(const v16bf*)(wbf + (size_t)(48 + l15) * C_IN + kw);

    acc0 = __builtin_amdgcn_wmma_f32_16x16x32_bf16(false, afrag, false, b0, (short)0, acc0, false, false);
    acc1 = __builtin_amdgcn_wmma_f32_16x16x32_bf16(false, afrag, false, b1, (short)0, acc1, false, false);
    acc2 = __builtin_amdgcn_wmma_f32_16x16x32_bf16(false, afrag, false, b2, (short)0, acc2, false, false);
    acc3 = __builtin_amdgcn_wmma_f32_16x16x32_bf16(false, afrag, false, b3, (short)0, acc3, false, false);
  }

  // Epilogue: SiLU then spatial mean. Each lane's 8 accum rows share column
  // n = tile*16 + l15 (C layout: lanes 0-15 M=j, lanes 16-31 M=j+8).
  float s0 = 0.f, s1 = 0.f, s2 = 0.f, s3 = 0.f;
  #pragma unroll
  for (int j = 0; j < 8; ++j) {
    s0 += silu_f(acc0[j]); s1 += silu_f(acc1[j]);
    s2 += silu_f(acc2[j]); s3 += silu_f(acc3[j]);
  }
  atomicAdd(&pool[ 0 + l15], s0);
  atomicAdd(&pool[16 + l15], s1);
  atomicAdd(&pool[32 + l15], s2);
  atomicAdd(&pool[48 + l15], s3);
  __syncthreads();

  const int b = blockIdx.x >> 1;        // 2 blocks (256 rows) per image
  if (tid < RED) atomicAdd(&pooled[b * RED + tid], pool[tid] * (1.0f / 256.0f));
}

// -------------------------------------------------------------------------
// Kernel 5: router head — logits, noisy top-2, softmax weights, aux loss
// Output layout: [0..63] weights, [64..127] indices(as float),
//                [128..383] logits, [384] aux_loss
// -------------------------------------------------------------------------
__global__ void __launch_bounds__(256) router_kernel(
    const float* __restrict__ pooled, const float* __restrict__ noise,
    const float* __restrict__ lin_w, const float* __restrict__ lin_b,
    float* __restrict__ out)
{
  __shared__ float lg[BATCH][NE];
  __shared__ float counts[NE];
  const int tid = threadIdx.x;
  const int b = tid >> 3, e = tid & 7;
  if (tid < NE) counts[tid] = 0.0f;

  float acc = lin_b[e];
  for (int r = 0; r < RED; ++r) acc += pooled[b * RED + r] * lin_w[e * RED + r];
  lg[b][e] = acc;
  out[128 + tid] = acc;                 // logits
  __syncthreads();

  if (e == 0) {
    float nl[NE];
    #pragma unroll
    for (int j = 0; j < NE; ++j) nl[j] = lg[b][j] + noise[b * NE + j];
    int i0 = 0;
    #pragma unroll
    for (int j = 1; j < NE; ++j) if (nl[j] > nl[i0]) i0 = j;   // ties -> lower idx
    int i1 = -1;
    #pragma unroll
    for (int j = 0; j < NE; ++j) {
      if (j == i0) continue;
      if (i1 < 0 || nl[j] > nl[i1]) i1 = j;
    }
    float r0 = lg[b][i0], r1 = lg[b][i1];
    float mx = fmaxf(r0, r1);
    float e0 = __expf(r0 - mx), e1 = __expf(r1 - mx);
    float inv = 1.0f / (e0 + e1);
    out[b * 2 + 0] = e0 * inv;
    out[b * 2 + 1] = e1 * inv;
    out[64 + b * 2 + 0] = (float)i0;
    out[64 + b * 2 + 1] = (float)i1;
    atomicAdd(&counts[i0], 1.0f);
    atomicAdd(&counts[i1], 1.0f);
  }
  __syncthreads();

  if (tid == 0) {
    float mean_prob[NE];
    for (int j = 0; j < NE; ++j) mean_prob[j] = 0.0f;
    for (int bb = 0; bb < BATCH; ++bb) {
      float mx = lg[bb][0];
      for (int j = 1; j < NE; ++j) mx = fmaxf(mx, lg[bb][j]);
      float s = 0.0f, ex[NE];
      for (int j = 0; j < NE; ++j) { ex[j] = __expf(lg[bb][j] - mx); s += ex[j]; }
      float inv = 1.0f / s;
      for (int j = 0; j < NE; ++j) mean_prob[j] += ex[j] * inv;
    }
    float loss = 0.0f;
    for (int j = 0; j < NE; ++j)
      loss += (counts[j] / (float)(BATCH * 2)) * (mean_prob[j] / (float)BATCH);
    out[384] = 2.0f * (float)NE * loss;
  }
}

// -------------------------------------------------------------------------
extern "C" void kernel_launch(void* const* d_in, const int* in_sizes, int n_in,
                              void* d_out, int out_size, void* d_ws, size_t ws_size,
                              hipStream_t stream) {
  const float* x     = (const float*)d_in[0];
  const float* noise = (const float*)d_in[1];
  const float* dwk   = (const float*)d_in[2];
  const float* gamma = (const float*)d_in[3];
  const float* beta  = (const float*)d_in[4];
  const float* pw    = (const float*)d_in[5];
  const float* lw    = (const float*)d_in[6];
  const float* lb    = (const float*)d_in[7];
  float* out = (float*)d_out;

  // workspace layout
  float* h      = (float*)d_ws;                       // 8192*1024 fp32 (32 MB)
  float* ch_sum = h + (size_t)M_TOT * C_IN;           // 1024
  float* ch_sq  = ch_sum + C_IN;                      // 1024
  float* scale  = ch_sq  + C_IN;                      // 1024
  float* shift  = scale  + C_IN;                      // 1024
  float* pooled = shift  + C_IN;                      // 32*64
  unsigned short* wbf = (unsigned short*)(pooled + BATCH * RED);  // 64*1024 bf16

  // zero accumulators (ch_sum, ch_sq, scale, shift, pooled are contiguous)
  const int nz = 4 * C_IN + BATCH * RED;
  init_zero_kernel<<<(nz + 255) / 256, 256, 0, stream>>>(ch_sum, nz);

  dw_conv_kernel<<<BATCH * (C_IN / 8), 256, 0, stream>>>(x, dwk, h, ch_sum, ch_sq);
  bn_finalize_kernel<<<C_IN / 256, 256, 0, stream>>>(ch_sum, ch_sq, gamma, beta, scale, shift);
  pack_w_kernel<<<(RED * C_IN) / 256, 256, 0, stream>>>(pw, wbf);
  gemm_pool_kernel<<<M_TOT / 128, 256, 0, stream>>>(h, scale, shift, wbf, pooled);
  router_kernel<<<1, 256, 0, stream>>>(pooled, noise, lw, lb, out);
}
